// SGRUCell_89696097009958
// MI455X (gfx1250) — matrix operand order repaired and checked by reading, not compiled, not run on hardware
//
#include <hip/hip_runtime.h>
#include <hip/hip_bf16.h>

// ---------------------------------------------------------------------------
// SGRU cell with plastic synapses for MI455X (gfx1250, wave32, WMMA).
// All GEMMs use v_wmma_f32_16x16x32_bf16 (f32 accumulate); f32 -> bf16
// conversion happens while staging tiles into LDS so weights are read from
// HBM exactly once as f32. Wave tiles are 16x64 (4 or 8 WMMAs per K-step)
// to raise WMMA density per LDS gather.
// ---------------------------------------------------------------------------

typedef __attribute__((ext_vector_type(16))) __bf16 v16bf;
typedef __attribute__((ext_vector_type(8)))  float  v8f;

#define TSM  64      // block tile M
#define TSN  128     // block tile N
#define KS   32      // K step (= WMMA K)
#define PAD  2       // LDS padding (bf16 elems) to decorrelate banks
#define TAU_U 0.01f

__device__ __forceinline__ float sigmoidf_(float x) {
    return 1.0f / (1.0f + __expf(-x));
}

// Convert 8 consecutive f32 in global memory to 8 bf16 in LDS.
__device__ __forceinline__ void stage8(const float* __restrict__ g, __bf16* l) {
    float4 f0 = *(const float4*)g;
    float4 f1 = *(const float4*)(g + 4);
    l[0] = (__bf16)f0.x; l[1] = (__bf16)f0.y; l[2] = (__bf16)f0.z; l[3] = (__bf16)f0.w;
    l[4] = (__bf16)f1.x; l[5] = (__bf16)f1.y; l[6] = (__bf16)f1.z; l[7] = (__bf16)f1.w;
}

// ---- WMMA fragment gathers (per CDNA5 ISA 7.12.2 layouts, wave32) ----------

// A fragment from tile stored [m][k] (row-major over K). 16x32 bf16.
// lanes 0-15: M=lane, K in {0..7, 16..23}; lanes 16-31: M=lane-16, K in {8..15, 24..31}
__device__ __forceinline__ v16bf fragA_mk(const __bf16* tile, int row_base,
                                          int lane, int stride) {
    int m  = row_base + (lane & 15);
    int kb = (lane & 16) ? 8 : 0;
    const __bf16* rp = tile + m * stride;
    v16bf a;
#pragma unroll
    for (int i = 0; i < 8; ++i) {
        int k = ((i & 4) << 2) + kb + ((i & 3) << 1);   // {0,16} + kb + 2*(i%4)
        a[2 * i]     = rp[k];
        a[2 * i + 1] = rp[k + 1];
    }
    return a;
}

// B fragment from tile stored [n][k]. 32x16 bf16.
// lanes 0-15: N=lane, K in {0..7, 16..23}; lanes 16-31: N=lane-16, K in {8..15, 24..31}
__device__ __forceinline__ v16bf fragB_nk(const __bf16* tile, int col_base,
                                          int lane, int stride) {
    int n  = col_base + (lane & 15);
    int kb = (lane & 16) ? 8 : 0;
    const __bf16* rp = tile + n * stride + kb;
    v16bf b;
#pragma unroll
    for (int i = 0; i < 8; ++i) {
        b[2 * i]     = rp[((i & 4) << 2) + ((i & 3) << 1)];
        b[2 * i + 1] = rp[((i & 4) << 2) + ((i & 3) << 1) + 1];
    }
    return b;
}

// A fragment from tile stored [k][c] (K-major): A[m][k] = tile[k][col_base+m]
__device__ __forceinline__ v16bf fragA_km(const __bf16* tile, int col_base,
                                          int lane, int stride) {
    int m  = col_base + (lane & 15);
    int kb = (lane & 16) ? 8 : 0;
    v16bf a;
#pragma unroll
    for (int i = 0; i < 8; ++i) {
        int k = ((i & 4) << 2) + kb + ((i & 3) << 1);
        a[2 * i]     = tile[k * stride + m];
        a[2 * i + 1] = tile[(k + 1) * stride + m];
    }
    return a;
}

// B fragment from tile stored [k][c]: B[k][n] = tile[k][col_base+n]
__device__ __forceinline__ v16bf fragB_kn(const __bf16* tile, int col_base,
                                          int lane, int stride) {
    int n  = col_base + (lane & 15);
    int kb = (lane & 16) ? 8 : 0;
    v16bf b;
#pragma unroll
    for (int i = 0; i < 8; ++i) {
        b[2 * i]     = tile[(((i & 4) << 2) + kb + ((i & 3) << 1)) * stride + n];
        b[2 * i + 1] = tile[(((i & 4) << 2) + kb + ((i & 3) << 1) + 1) * stride + n];
    }
    return b;
}

// ---------------------------------------------------------------------------
// C[M,N] = A[M,K] * B[N,K]^T (+ bias[N]); A,B f32 in HBM, bf16 in LDS.
// Block: 256 threads = 8 waves; tile 64x128; wave (wm,wn) computes 16x64
// via 4 back-to-back WMMAs per K-step.
// ---------------------------------------------------------------------------
__global__ __launch_bounds__(256) void gemm_abT_bf16(
    const float* __restrict__ A, const float* __restrict__ B,
    const float* __restrict__ bias, float* __restrict__ C,
    int K, int lda, int ldb, int ldc)
{
    __shared__ __bf16 As[TSM][KS + PAD];
    __shared__ __bf16 Bs[TSN][KS + PAD];

    const int tid  = threadIdx.x;
    const int lane = tid & 31;
    const int w    = tid >> 5;
    const int wm   = (w & 3) << 4;     // 0,16,32,48
    const int wn   = (w >> 2) << 6;    // 0,64
    const int m0   = blockIdx.x * TSM;
    const int n0   = blockIdx.y * TSN;

    // A staging: 64x32 f32 = 2048 elems, 8 per thread
    const int lrA = tid >> 2;          // 0..63
    const int lcA = (tid & 3) << 3;    // 0,8,16,24
    // B staging: 128x32 f32 = 4096 elems, 16 per thread
    const int lrB = tid >> 1;          // 0..127
    const int lcB = (tid & 1) << 4;    // 0,16

    v8f acc[4];
#pragma unroll
    for (int jn = 0; jn < 4; ++jn)
        acc[jn] = (v8f){0.f,0.f,0.f,0.f,0.f,0.f,0.f,0.f};

    const float* aRow = A + (size_t)(m0 + lrA) * lda + lcA;
    const float* bRow = B + (size_t)(n0 + lrB) * ldb + lcB;

    for (int k0 = 0; k0 < K; k0 += KS) {
        __syncthreads();
        stage8(aRow + k0,     &As[lrA][lcA]);
        stage8(bRow + k0,     &Bs[lrB][lcB]);
        stage8(bRow + k0 + 8, &Bs[lrB][lcB + 8]);
        if (k0 + KS < K) {             // hint next K tile (global_prefetch_b8)
            __builtin_prefetch(aRow + k0 + KS, 0, 3);
            __builtin_prefetch(bRow + k0 + KS, 0, 3);
        }
        __syncthreads();

        v16bf af = fragA_mk(&As[0][0], wm, lane, KS + PAD);
#pragma unroll
        for (int jn = 0; jn < 4; ++jn) {
            v16bf bf = fragB_nk(&Bs[0][0], wn + (jn << 4), lane, KS + PAD);
            acc[jn] = __builtin_amdgcn_wmma_f32_16x16x32_bf16(
                          false, af, false, bf, (short)0, acc[jn], false, false);
        }
    }

    // C/D layout: VGPR j -> M = j (lanes 0-15) or j+8 (lanes 16-31); N = lane&15
    const int nc   = lane & 15;
    const int mrow = (lane & 16) ? 8 : 0;
#pragma unroll
    for (int jn = 0; jn < 4; ++jn) {
        const int ncol = n0 + wn + (jn << 4) + nc;
        const float bb = bias ? bias[ncol] : 0.f;
#pragma unroll
        for (int j = 0; j < 8; ++j) {
            C[(size_t)(m0 + wm + mrow + j) * ldc + ncol] = acc[jn][j] + bb;
        }
    }
}

// ---------------------------------------------------------------------------
// Fused gate / plasticity / spiking elementwise pass over [B,H].
// ---------------------------------------------------------------------------
__global__ __launch_bounds__(256) void gates_kernel(
    const float* __restrict__ Wx, const float* __restrict__ Wh,
    const float* __restrict__ plastic, const float* __restrict__ v,
    const float* __restrict__ trace, const float* __restrict__ alpha,
    float* __restrict__ out_v, float* __restrict__ out_h,
    float* __restrict__ out_trace, int Bdim, int Hdim)
{
    int idx = blockIdx.x * blockDim.x + threadIdx.x;
    if (idx >= Bdim * Hdim) return;
    int b = idx / Hdim;
    int n = idx - b * Hdim;
    size_t row = (size_t)b * 3 * Hdim;

    float z  = sigmoidf_(Wx[row + n]        + Wh[row + n]);
    float r  = sigmoidf_(Wx[row + Hdim + n] + Wh[row + Hdim + n]);
    float dv = Wx[row + 2 * Hdim + n] +
               r * (Wh[row + 2 * Hdim + n] + fabsf(alpha[n]) * plastic[idx]);
    float nv = (1.0f - z) * v[idx] + z * dv;
    float re = nv > 0.0f ? nv : 0.0f;
    float nh = 1.0f - __expf(-re);

    out_v[idx]     = nv;
    out_h[idx]     = nh;
    out_trace[idx] = (1.0f - z) * trace[idx] + z * r * nh;
}

// ---------------------------------------------------------------------------
// new_dU[i,j] = clip((1-tau)*dU[i,j]
//                    + tau * (sum_b nh[b,i]*tr[b,j] - tr[b,i]*nh[b,j]), -1, 1)
// A^T*B GEMMs, K = B = 512. Tiles staged K-major [k][col], bf16.
// Block tile 64(i) x 128(j); each wave 16x64 for BOTH products -> 8 WMMAs.
// ---------------------------------------------------------------------------
__global__ __launch_bounds__(256) void gemm_stdp(
    const float* __restrict__ NH, const float* __restrict__ TR,
    const float* __restrict__ dU, float* __restrict__ dU_out,
    int Hdim, int K)
{
    __shared__ __bf16 nhM[KS][TSM + PAD];
    __shared__ __bf16 trM[KS][TSM + PAD];
    __shared__ __bf16 nhN[KS][TSN + PAD];
    __shared__ __bf16 trN[KS][TSN + PAD];

    const int tid  = threadIdx.x;
    const int lane = tid & 31;
    const int w    = tid >> 5;
    const int wm   = (w & 3) << 4;     // 0..48
    const int wn   = (w >> 2) << 6;    // 0,64
    const int i0   = blockIdx.x * TSM;
    const int j0   = blockIdx.y * TSN;

    // M-tiles: 32x64 f32 = 2048 elems, 8 per thread
    const int krM = tid >> 3;          // 0..31
    const int ccM = (tid & 7) << 3;    // 0..56
    // N-tiles: 32x128 f32 = 4096 elems, 16 per thread (two rows)
    const int krN = tid >> 4;          // 0..15 (and +16)
    const int ccN = (tid & 15) << 3;   // 0..120

    v8f acc1[4], acc2[4];
#pragma unroll
    for (int jn = 0; jn < 4; ++jn) {
        acc1[jn] = (v8f){0.f,0.f,0.f,0.f,0.f,0.f,0.f,0.f};
        acc2[jn] = (v8f){0.f,0.f,0.f,0.f,0.f,0.f,0.f,0.f};
    }

    for (int k0 = 0; k0 < K; k0 += KS) {
        __syncthreads();
        {
            const size_t grM = (size_t)(k0 + krM) * Hdim;
            stage8(NH + grM + i0 + ccM, &nhM[krM][ccM]);
            stage8(TR + grM + i0 + ccM, &trM[krM][ccM]);
            const size_t grN0 = (size_t)(k0 + krN) * Hdim;
            const size_t grN1 = (size_t)(k0 + krN + 16) * Hdim;
            stage8(NH + grN0 + j0 + ccN, &nhN[krN][ccN]);
            stage8(NH + grN1 + j0 + ccN, &nhN[krN + 16][ccN]);
            stage8(TR + grN0 + j0 + ccN, &trN[krN][ccN]);
            stage8(TR + grN1 + j0 + ccN, &trN[krN + 16][ccN]);
        }
        __syncthreads();

        v16bf fa1 = fragA_km(&nhM[0][0], wm, lane, TSM + PAD);
        v16bf fa2 = fragA_km(&trM[0][0], wm, lane, TSM + PAD);
#pragma unroll
        for (int jn = 0; jn < 4; ++jn) {
            v16bf b1 = fragB_kn(&trN[0][0], wn + (jn << 4), lane, TSN + PAD);
            v16bf b2 = fragB_kn(&nhN[0][0], wn + (jn << 4), lane, TSN + PAD);
            acc1[jn] = __builtin_amdgcn_wmma_f32_16x16x32_bf16(
                           false, fa1, false, b1, (short)0, acc1[jn], false, false);
            acc2[jn] = __builtin_amdgcn_wmma_f32_16x16x32_bf16(
                           false, fa2, false, b2, (short)0, acc2[jn], false, false);
        }
    }

    const int nc   = lane & 15;
    const int mrow = (lane & 16) ? 8 : 0;
#pragma unroll
    for (int jn = 0; jn < 4; ++jn) {
        const int jcol = j0 + wn + (jn << 4) + nc;
#pragma unroll
        for (int j = 0; j < 8; ++j) {
            size_t idx = (size_t)(i0 + wm + mrow + j) * Hdim + jcol;
            float dd  = acc1[jn][j] - acc2[jn][j];
            float val = (1.0f - TAU_U) * dU[idx] + TAU_U * dd;
            dU_out[idx] = fminf(fmaxf(val, -1.0f), 1.0f);
        }
    }
}

// ---------------------------------------------------------------------------
extern "C" void kernel_launch(void* const* d_in, const int* in_sizes, int n_in,
                              void* d_out, int out_size, void* d_ws, size_t ws_size,
                              hipStream_t stream) {
    (void)in_sizes; (void)n_in; (void)out_size; (void)ws_size;

    const float* x     = (const float*)d_in[0];   // [B, D]
    const float* h     = (const float*)d_in[1];   // [B, H]
    const float* v     = (const float*)d_in[2];   // [B, H]
    const float* dU    = (const float*)d_in[3];   // [H, H]
    const float* trace = (const float*)d_in[4];   // [B, H]
    const float* Wx_w  = (const float*)d_in[5];   // [3H, D]
    const float* Wx_b  = (const float*)d_in[6];   // [3H]
    const float* Wh_w  = (const float*)d_in[7];   // [3H, H]
    const float* Wh_b  = (const float*)d_in[8];   // [3H]
    const float* alpha = (const float*)d_in[9];   // [1, H]

    const int Bn = 512, Dn = 1024, Hn = 2048;
    const int H3 = 3 * Hn;

    float* out_v  = (float*)d_out;                    // [B, H]
    float* out_h  = out_v  + (size_t)Bn * Hn;         // [B, H]
    float* out_dU = out_h  + (size_t)Bn * Hn;         // [H, H]
    float* out_tr = out_dU + (size_t)Hn * Hn;         // [B, H]

    // Workspace: Wx [B,3H] + Wh [B,3H] + plastic [B,H]  (~28 MB f32)
    float* Wx      = (float*)d_ws;
    float* Wh      = Wx + (size_t)Bn * H3;
    float* plastic = Wh + (size_t)Bn * H3;

    dim3 blk(256);

    // Wx = x @ Wx_w^T + Wx_b   : M=512, N=6144, K=1024
    gemm_abT_bf16<<<dim3(Bn / TSM, H3 / TSN), blk, 0, stream>>>(
        x, Wx_w, Wx_b, Wx, Dn, Dn, Dn, H3);

    // Wh = h @ Wh_w^T + Wh_b   : M=512, N=6144, K=2048
    gemm_abT_bf16<<<dim3(Bn / TSM, H3 / TSN), blk, 0, stream>>>(
        h, Wh_w, Wh_b, Wh, Hn, Hn, Hn, H3);

    // plastic = h @ dU^T        : M=512, N=2048, K=2048
    gemm_abT_bf16<<<dim3(Bn / TSM, Hn / TSN), blk, 0, stream>>>(
        h, dU, nullptr, plastic, Hn, Hn, Hn, Hn);

    // Fused gates / membrane / spiking / trace update
    gates_kernel<<<(Bn * Hn + 255) / 256, blk, 0, stream>>>(
        Wx, Wh, plastic, v, trace, alpha, out_v, out_h, out_tr, Bn, Hn);

    // STDP antisymmetric update, fused decay + clip
    gemm_stdp<<<dim3(Hn / TSM, Hn / TSN), blk, 0, stream>>>(
        out_h, trace, dU, out_dU, Hn, Bn);
}
